// GaussianLTC_54803782697416
// MI455X (gfx1250) — compile-verified, matrix operand order checked by reading, not run on hardware
//
#include <hip/hip_runtime.h>
#include <math.h>

// Problem constants (match reference)
#define B_  64
#define T_  128
#define H_  128      // F == H == 128
#define L_  2
#define UNF 6
#define EPS_ 1e-8f
#define HH  (H_*H_)        // 16384
#define KTOT (T_*H_)       // 16384 flattened hidden per batch

typedef __attribute__((ext_vector_type(16))) _Float16 v16h;
typedef __attribute__((ext_vector_type(8)))  float    v8f;
typedef __attribute__((ext_vector_type(4)))  _Float16 h4;    // 8-byte packed params
typedef __attribute__((ext_vector_type(4)))  unsigned int u32x4;
typedef __attribute__((ext_vector_type(8)))  int i32x8;
typedef __attribute__((ext_vector_type(4)))  int i32x4;

__device__ __forceinline__ float sigmoidf_(float x) {
    return 1.0f / (1.0f + __expf(-x));
}

// ---------------------------------------------------------------------------
// Prep: fold and PACK parameters as half4 {sigma, mu*sigma, w, w*erev} so each
// layer's recurrent + sensory synapse tables are 2 x 128KB -> both fit in the
// 320KB CDNA5 LDS. Workspace layout per layer l (h4 units):
//   [l*2*HH + 0  .. HH)   recurrent pack
//   [l*2*HH + HH .. 2HH)  sensory pack
// ---------------------------------------------------------------------------
__global__ void ltc_pack_kernel(const float* __restrict__ sigma, const float* __restrict__ mu,
                                const float* __restrict__ w,     const float* __restrict__ erev,
                                const float* __restrict__ ssig,  const float* __restrict__ smu,
                                const float* __restrict__ sw,    const float* __restrict__ serev,
                                h4* __restrict__ pack) {
    int i = blockIdx.x * blockDim.x + threadIdx.x;
    if (i < L_ * HH) {
        const int l = i >> 14;          // / HH
        const int r = i & (HH - 1);
        h4 rp, sp;
        rp.x = (_Float16)sigma[i];
        rp.y = (_Float16)(mu[i] * sigma[i]);
        rp.z = (_Float16)w[i];
        rp.w = (_Float16)(w[i] * erev[i]);
        sp.x = (_Float16)ssig[i];
        sp.y = (_Float16)(smu[i] * ssig[i]);
        sp.z = (_Float16)sw[i];
        sp.w = (_Float16)(sw[i] * serev[i]);
        pack[(size_t)l * 2 * HH + r]      = rp;
        pack[(size_t)l * 2 * HH + HH + r] = sp;
    }
}

// ---------------------------------------------------------------------------
// One LTC layer. grid = B (one block / WGP), block = 512 = (j 0..127) x (ig 0..3).
// All synapse params live in LDS (256KB, staged once via the Tensor Data Mover).
// Per-timestep global traffic: one 512B input row. Dynamic LDS total: 268,288B.
// ---------------------------------------------------------------------------
__global__ __launch_bounds__(512, 1)
void ltc_layer_kernel(const float* __restrict__ in_seq,   // [B, T, 128]
                      float* __restrict__ out_seq,        // [B, T, 128]
                      const float* __restrict__ gl,       // [H] (layer slice)
                      const float* __restrict__ vl,       // [H]
                      const float* __restrict__ cmv,      // [H]
                      const h4*    __restrict__ packL)    // [2*HH] rec then sen
{
    extern __shared__ float smem[];
    h4*    pRec = (h4*)smem;            // 16384 * 8B = 128KB
    h4*    pSen = pRec + HH;            // 128KB
    float* vsh  = (float*)(pSen + HH);  // 128 membrane state
    float* ush  = vsh  + H_;            // 128 input row
    float* numS = ush  + H_;            // 128 sensory numerator
    float* denS = numS + H_;            // 128 sensory denominator
    float* redN = denS + H_;            // 4*128 partial reduction
    float* redD = redN + 4*H_;          // 4*128

    const int tid = threadIdx.x;
    const int b   = blockIdx.x;
    const int j   = tid & (H_ - 1);     // output neuron
    const int ig  = tid >> 7;           // 0..3 : i-reduction slice
    const int i0  = ig * 32;

#if __has_builtin(__builtin_amdgcn_tensor_load_to_lds)
    // ---- Stage both packed tables (256KB) with one TDM descriptor ----------
    // D# group0: count=1, lds_addr, global_addr, type=2 (ISA 8.3)
    // D# group1: data_size=4B, tensor/tile = 32768 x 2 dwords (ISA 8.4)
    if (tid < 32) {   // wave 0 only: TENSORcnt is per-wave
        const unsigned long long ga = (unsigned long long)(uintptr_t)packL;
        const unsigned int lds_base = (unsigned int)(uintptr_t)pRec;
        u32x4 g0;
        g0[0] = 1u;                                            // count=1
        g0[1] = lds_base;                                      // lds_addr
        g0[2] = (unsigned int)(ga & 0xFFFFFFFFu);              // global_addr lo
        g0[3] = (unsigned int)((ga >> 32) & 0x01FFFFFFu) | 0x80000000u; // hi | type=2
        i32x8 g1;
        g1[0] = 0x00020000;        // wg_mask=0, data_size=2 (4B)
        g1[1] = (int)0x80000000;   // tensor_dim0 = 32768 (lo16 in [31:16])
        g1[2] = 0x00020000;        // tensor_dim0 hi=0, tensor_dim1 = 2
        g1[3] = (int)0x80000000;   // tensor_dim1 hi=0, tile_dim0 = 32768
        g1[4] = 0x00000002;        // tile_dim1 = 2, tile_dim2 = 0
        g1[5] = 0x00008000;        // tensor_dim0_stride = 32768
        g1[6] = 0;                 // stride hi / dim1_stride lo
        g1[7] = 0;
        i32x4 gz  = {0, 0, 0, 0};
        i32x8 gz8 = {0, 0, 0, 0, 0, 0, 0, 0};
        // amdgpu-toolchain (clang-23) 6-arg form.
        __builtin_amdgcn_tensor_load_to_lds(g0, g1, gz, gz, gz8, 0);
        __builtin_amdgcn_s_wait_tensorcnt(0);
    }
#else
    // ---- Fallback: cooperative vectorized copy ----------------------------
    for (int idx = tid; idx < 2 * HH; idx += blockDim.x)
        pRec[idx] = packL[idx];
#endif
    if (tid < H_) vsh[tid] = 0.0f;

    // Hoist per-neuron leak parameters (used only by tid < 128).
    float g = 0.f, vlk = 0.f, cmt = 0.f;
    if (tid < H_) {
        g   = gl[tid];
        vlk = vl[tid];
        cmt = cmv[tid] * (float)UNF;    // cm / (elapsed/unfolds)
    }
    __syncthreads();

    for (int t = 0; t < T_; ++t) {
        // Load this timestep's input row; prefetch next one (global_prefetch_b8).
        if (tid < H_) {
            ush[tid] = in_seq[(size_t)(b * T_ + t) * H_ + tid];
            if (t + 1 < T_)
                __builtin_prefetch(in_seq + (size_t)(b * T_ + t + 1) * H_ + tid, 0, 3);
        }
        __syncthreads();

        // ---- sensory synapses (once per t), params in LDS ----
        float an = 0.f, ad = 0.f;
        #pragma unroll 4
        for (int ii = 0; ii < 32; ++ii) {
            const int o = (i0 + ii) * H_ + j;          // ds_load_b64, conflict-free
            const h4 p = pSen[o];
            const float s = sigmoidf_(ush[i0 + ii] * (float)p.x - (float)p.y);
            ad += (float)p.z * s;
            an += (float)p.w * s;
        }
        redN[ig * H_ + j] = an;
        redD[ig * H_ + j] = ad;
        __syncthreads();
        if (tid < H_) {
            numS[tid] = redN[tid] + redN[H_ + tid] + redN[2*H_ + tid] + redN[3*H_ + tid];
            denS[tid] = redD[tid] + redD[H_ + tid] + redD[2*H_ + tid] + redD[3*H_ + tid];
        }
        __syncthreads();

        // ---- fused ODE unfolds, params in LDS ----
        for (int u = 0; u < UNF; ++u) {
            float bn = 0.f, bd = 0.f;
            #pragma unroll 4
            for (int ii = 0; ii < 32; ++ii) {
                const int o = (i0 + ii) * H_ + j;
                const h4 p = pRec[o];
                const float s = sigmoidf_(vsh[i0 + ii] * (float)p.x - (float)p.y);
                bd += (float)p.z * s;
                bn += (float)p.w * s;
            }
            redN[ig * H_ + j] = bn;
            redD[ig * H_ + j] = bd;
            __syncthreads();
            if (tid < H_) {
                const float num = redN[tid] + redN[H_+tid] + redN[2*H_+tid] + redN[3*H_+tid] + numS[tid];
                const float den = redD[tid] + redD[H_+tid] + redD[2*H_+tid] + redD[3*H_+tid] + denS[tid];
                const float vp  = vsh[tid];
                vsh[tid] = (cmt * vp + g * vlk + num) / (cmt + g + den + EPS_);
            }
            __syncthreads();
        }

        if (tid < H_) out_seq[(size_t)(b * T_ + t) * H_ + tid] = vsh[tid];
        __syncthreads();
    }
}

// ---------------------------------------------------------------------------
// Dual linear heads as one WMMA GEMM: C[64x16] = h[64x16384] x Wc^T, where
// Wc rows 0..1 = mu_w, 2..3 = ls_w, 4..15 = zero. One wave per 16-batch tile.
// v_wmma_f32_16x16x32_f16 with ISA 16-bit A/B layouts:
// lane l holds row/col l%16, element K(v,e) = (v>=4)*16 + (l/16)*8 + (v%4)*2 + e.
// ---------------------------------------------------------------------------
__global__ __launch_bounds__(128, 1)
void head_wmma_kernel(const float* __restrict__ h,        // [64, 16384]
                      const float* __restrict__ mu_w,     // [2, 16384]
                      const float* __restrict__ mu_b,     // [2]
                      const float* __restrict__ ls_w,     // [2, 16384]
                      const float* __restrict__ ls_b,     // [2]
                      float* __restrict__ out)            // [256]: mu[128], sigma[128]
{
    const int tid  = threadIdx.x;
    const int wave = tid >> 5;           // 0..3 -> batch tile
    const int lane = tid & 31;
    const int m16  = lane & 15;
    const int half = lane >> 4;

    const int  brow = wave * 16 + m16;                    // A row (batch)
    const int  ncol = m16;                                // B col (head)
    // Uniform control flow: EXEC stays all-1s around the WMMA (ISA requirement).
    const float* wrow  = (ncol < 2) ? (mu_w + (size_t)ncol * KTOT)
                        : (ncol < 4) ? (ls_w + (size_t)(ncol - 2) * KTOT)
                                     : mu_w;
    const float  wmask = (ncol < 4) ? 1.0f : 0.0f;
    const float* hrow  = h + (size_t)brow * KTOT;

    v8f c = {};
    for (int kb = 0; kb < KTOT; kb += 32) {
        v16h a, bm;
        #pragma unroll
        for (int v = 0; v < 8; ++v) {
            const int k0 = ((v >> 2) << 4) + (half << 3) + ((v & 3) << 1);
            const float2 ha = *(const float2*)(hrow + kb + k0);
            a[2*v]   = (_Float16)ha.x;
            a[2*v+1] = (_Float16)ha.y;
            const float2 wv = *(const float2*)(wrow + kb + k0);
            bm[2*v]   = (_Float16)(wv.x * wmask);
            bm[2*v+1] = (_Float16)(wv.y * wmask);
        }
        // D = A(16x32 f16) * B(32x16 f16) + C(f32)
        c = __builtin_amdgcn_wmma_f32_16x16x32_f16(false, a, false, bm,
                                                   (short)0, c, false, false);
    }

    // C/D layout: lane l, VGPR r -> M = r + 8*(l/16), N = l%16.
    #pragma unroll
    for (int r = 0; r < 8; ++r) {
        const int M  = r + 8 * half;
        const int bo = wave * 16 + M;
        if (ncol < 2)        out[bo * 2 + ncol]            = c[r] + mu_b[ncol];
        else if (ncol < 4)   out[128 + bo * 2 + (ncol-2)]  = __expf(c[r] + ls_b[ncol-2]);
    }
}

// ---------------------------------------------------------------------------
extern "C" void kernel_launch(void* const* d_in, const int* in_sizes, int n_in,
                              void* d_out, int out_size, void* d_ws, size_t ws_size,
                              hipStream_t stream) {
    const float* x       = (const float*)d_in[0];   // [B,T,F]
    const float* gleak   = (const float*)d_in[1];   // [L,H]
    const float* vleak   = (const float*)d_in[2];
    const float* cm      = (const float*)d_in[3];
    const float* sigma   = (const float*)d_in[4];   // [L,H,H]
    const float* mu_syn  = (const float*)d_in[5];
    const float* w       = (const float*)d_in[6];
    const float* erev    = (const float*)d_in[7];
    const float* s_sigma = (const float*)d_in[8];
    const float* s_mu    = (const float*)d_in[9];
    const float* s_w     = (const float*)d_in[10];
    const float* s_erev  = (const float*)d_in[11];
    const float* mu_w    = (const float*)d_in[12];  // [LAT, T*H]
    const float* mu_b    = (const float*)d_in[13];
    const float* ls_w    = (const float*)d_in[14];
    const float* ls_b    = (const float*)d_in[15];

    // Workspace: packed params (2*L*HH h4 = 512KB) then two hidden traces (4MB each).
    h4*    pack = (h4*)d_ws;
    float* hs0  = (float*)(pack + (size_t)2 * L_ * HH);   // [B,T,H]
    float* hs1  = hs0 + (size_t)B_ * T_ * H_;             // [B,T,H]

    // 1) Fold + pack parameters to half4.
    ltc_pack_kernel<<<(L_ * HH + 255) / 256, 256, 0, stream>>>(
        sigma, mu_syn, w, erev, s_sigma, s_mu, s_w, s_erev, pack);

    // 2) Two LTC layers (sequential on stream). 268,288B dynamic LDS per block;
    //    CDNA5 WGP provides 320KB (one block per WGP).
    const size_t shmem = (size_t)(2 * HH) * sizeof(h4) + (size_t)(12 * H_) * sizeof(float);
    ltc_layer_kernel<<<B_, 512, shmem, stream>>>(
        x, hs0, gleak, vleak, cm, pack);
    ltc_layer_kernel<<<B_, 512, shmem, stream>>>(
        hs0, hs1, gleak + H_, vleak + H_, cm + H_, pack + (size_t)2 * HH);

    // 3) WMMA heads -> d_out[0..127] = mu, d_out[128..255] = exp(log-sigma).
    head_wmma_kernel<<<1, 128, 0, stream>>>(hs1, mu_w, mu_b, ls_w, ls_b, (float*)d_out);
}
